// Attention_5935644803277
// MI455X (gfx1250) — matrix operand order repaired and checked by reading, not compiled, not run on hardware
//
#include <hip/hip_runtime.h>

typedef _Float16 half_t;
typedef __attribute__((ext_vector_type(16))) _Float16 v16h;
typedef __attribute__((ext_vector_type(8)))  _Float16 v8h;
typedef __attribute__((ext_vector_type(8)))  float    v8f;

#define NB    32
#define DIM   768
#define NHD   12
#define HDD   64
#define LQ    1536
#define LKV   384
#define MTKV  128
#define EPS   1e-5f
// scale = 768^-0.5 (reference scales by full D, not head dim)
#define QK_SCALE 0.03608439182435161f

// ---------------------------------------------------------------------------
// fp32 -> f16 conversion (weights)
// ---------------------------------------------------------------------------
__global__ void cvt_f32_f16(const float* __restrict__ in, half_t* __restrict__ out, int n) {
    int i = blockIdx.x * blockDim.x + threadIdx.x;
    if (i < n) out[i] = (half_t)in[i];
}

// ---------------------------------------------------------------------------
// Depthwise 3x3 conv + BN, stride 1 (Q path). One block per (batch, token).
// ---------------------------------------------------------------------------
__global__ void convbn_s1(const float* __restrict__ x,
                          const float* __restrict__ w,
                          const float* __restrict__ g,  const float* __restrict__ bb,
                          const float* __restrict__ mn, const float* __restrict__ vr,
                          half_t* __restrict__ out) {
    int idx = blockIdx.x;
    int b = idx / LQ;
    int t = idx % LQ;
    int base, H, W, p;
    if (t < 256)      { base = 0;   H = 16; W = 16; p = t; }
    else if (t < 512) { base = 256; H = 16; W = 16; p = t - 256; }
    else              { base = 512; H = 32; W = 32; p = t - 512; }
    int h = p / W, wp = p % W;
    const float* xb = x + (size_t)b * LQ * DIM;
    for (int d = threadIdx.x; d < DIM; d += blockDim.x) {
        float acc = 0.0f;
        #pragma unroll
        for (int kh = 0; kh < 3; ++kh) {
            int ih = h + kh - 1;
            if (ih < 0 || ih >= H) continue;
            #pragma unroll
            for (int kw = 0; kw < 3; ++kw) {
                int iw = wp + kw - 1;
                if (iw < 0 || iw >= W) continue;
                acc += xb[(size_t)(base + ih * W + iw) * DIM + d] * w[d * 9 + kh * 3 + kw];
            }
        }
        float inv = g[d] * rsqrtf(vr[d] + EPS);
        out[((size_t)b * LQ + t) * DIM + d] = (half_t)(acc * inv + (bb[d] - mn[d] * inv));
    }
}

// ---------------------------------------------------------------------------
// Depthwise 3x3 conv + BN, stride 2 (K and V in one pass; shares input reads).
// ---------------------------------------------------------------------------
__global__ void convbn_s2(const float* __restrict__ x,
                          const float* __restrict__ wk, const float* __restrict__ gk,
                          const float* __restrict__ bk, const float* __restrict__ mk,
                          const float* __restrict__ vk,
                          const float* __restrict__ wv, const float* __restrict__ gv,
                          const float* __restrict__ bv, const float* __restrict__ mv,
                          const float* __restrict__ vv,
                          half_t* __restrict__ outk, half_t* __restrict__ outv) {
    int idx = blockIdx.x;
    int b = idx / LKV;
    int t = idx % LKV;
    int base, H, W, OW, p;
    if (t < 64)       { base = 0;   H = 16; W = 16; OW = 8;  p = t; }
    else if (t < 128) { base = 256; H = 16; W = 16; OW = 8;  p = t - 64; }
    else              { base = 512; H = 32; W = 32; OW = 16; p = t - 128; }
    int oh = p / OW, ow = p % OW;
    const float* xb = x + (size_t)b * LQ * DIM;
    for (int d = threadIdx.x; d < DIM; d += blockDim.x) {
        float ak = 0.0f, av = 0.0f;
        #pragma unroll
        for (int kh = 0; kh < 3; ++kh) {
            int ih = 2 * oh + kh - 1;
            if (ih < 0 || ih >= H) continue;
            #pragma unroll
            for (int kw = 0; kw < 3; ++kw) {
                int iw = 2 * ow + kw - 1;
                if (iw < 0 || iw >= W) continue;
                float xv = xb[(size_t)(base + ih * W + iw) * DIM + d];
                ak += xv * wk[d * 9 + kh * 3 + kw];
                av += xv * wv[d * 9 + kh * 3 + kw];
            }
        }
        float ik = gk[d] * rsqrtf(vk[d] + EPS);
        float iv = gv[d] * rsqrtf(vv[d] + EPS);
        outk[((size_t)b * LKV + t) * DIM + d] = (half_t)(ak * ik + (bk[d] - mk[d] * ik));
        outv[((size_t)b * LKV + t) * DIM + d] = (half_t)(av * iv + (bv[d] - mv[d] * iv));
    }
}

// ---------------------------------------------------------------------------
// WMMA A-operand fragment (16x32 f16): lane m=l&15, kb=(l>>4)*8,
// halves [0..15]=K 2*kb..2*kb+15 (contiguous, matches B fragments)
// ---------------------------------------------------------------------------
__device__ __forceinline__ v16h make_a_frag(const half_t* __restrict__ rowp, int kb) {
    v8h lo = *reinterpret_cast<const v8h*>(rowp + 2 * kb);
    v8h hi = *reinterpret_cast<const v8h*>(rowp + 2 * kb + 8);
    v16h a;
    #pragma unroll
    for (int i = 0; i < 8; ++i) { a[i] = lo[i]; a[i + 8] = hi[i]; }
    return a;
}

// ---------------------------------------------------------------------------
// GEMM: C(M x 768) = A(M x 768, f16) * W(768 x 768, K-major)^T
// Per-wave 16x64 tile, software-pipelined k-loop (prefetch k+32 fragments
// before the current WMMAs so VMEM overlaps the matrix pipe). Output goes to
// per-head layout [b, h, t, 64].
// ---------------------------------------------------------------------------
__global__ void gemm_qkv(const half_t* __restrict__ A, const half_t* __restrict__ Wt,
                         half_t* __restrict__ outH, int M, int T) {
    const int NG = DIM / 64;
    int wave = blockIdx.x * (blockDim.x >> 5) + (threadIdx.x >> 5);
    int lane = threadIdx.x & 31;
    int mt = wave / NG, ng = wave % NG;
    if (mt >= M / 16) return;
    int row0 = mt * 16, n0 = ng * 64;
    int mloc = lane & 15, kb = (lane >> 4) * 8, koff = (lane >> 4) * 16;
    const half_t* arow = A + (size_t)(row0 + mloc) * DIM;
    const half_t* wrow[4];
    #pragma unroll
    for (int nt = 0; nt < 4; ++nt)
        wrow[nt] = Wt + (size_t)(n0 + nt * 16 + (lane & 15)) * DIM + koff;

    v8f acc[4] = {};
    v16h a_c = make_a_frag(arow, kb);
    v16h b_c[4];
    #pragma unroll
    for (int nt = 0; nt < 4; ++nt)
        b_c[nt] = *reinterpret_cast<const v16h*>(wrow[nt]);

    for (int k0 = 0; k0 < DIM; k0 += 32) {
        int kn = (k0 + 32 < DIM) ? (k0 + 32) : k0;   // clamp: no branch, no divergence
        v16h a_n = make_a_frag(arow + kn, kb);
        v16h b_n[4];
        #pragma unroll
        for (int nt = 0; nt < 4; ++nt)
            b_n[nt] = *reinterpret_cast<const v16h*>(wrow[nt] + kn);
        #pragma unroll
        for (int nt = 0; nt < 4; ++nt)
            acc[nt] = __builtin_amdgcn_wmma_f32_16x16x32_f16(
                false, a_c, false, b_c[nt], (short)0, acc[nt], false, false);
        a_c = a_n;
        #pragma unroll
        for (int nt = 0; nt < 4; ++nt) b_c[nt] = b_n[nt];
    }
    int g2 = (lane >> 4) * 8;
    #pragma unroll
    for (int nt = 0; nt < 4; ++nt) {
        #pragma unroll
        for (int r = 0; r < 8; ++r) {
            int row = row0 + r + g2;
            int col = n0 + nt * 16 + (lane & 15);
            int bidx = row / T, tt = row % T;
            int hh = col >> 6, dd = col & 63;
            outH[(((size_t)bidx * NHD + hh) * T + tt) * HDD + dd] = (half_t)acc[nt][r];
        }
    }
}

// ---------------------------------------------------------------------------
// Flash attention. Block = 4 waves sharing one (b,h); per 32-kv block:
//   - K staged to LDS with gfx1250 async copy (global_load_async_to_lds_b128,
//     ASYNCcnt) -- linear 4KB, B-operand reads become contiguous DS loads
//   - V staged TRANSPOSED to LDS so the P*V B-fragment is one contiguous
//     32B DS read per d-tile (replaces 64 scalar global loads per wave)
//   - online softmax, P routed through a per-wave LDS slab (C/D -> A layout)
// ---------------------------------------------------------------------------
__global__ void attn_flash(const half_t* __restrict__ Qh, const half_t* __restrict__ Kh,
                           const half_t* __restrict__ Vh, half_t* __restrict__ xattn,
                           int qoff, int Tkv) {
    __shared__ _Float16 Kbuf[32][64];      // [kv][d]   4KB
    __shared__ _Float16 Vt[64][32];        // [d][kv]   4KB (transposed)
    __shared__ _Float16 Plds[4][16][32];   // per-wave P slabs 4KB
    int tid = threadIdx.x;
    int wid = tid >> 5, lane = tid & 31;
    int bh = blockIdx.x;
    int bq = bh / NHD, hh = bh % NHD;
    int qrow0 = qoff + (blockIdx.y * 4 + wid) * 16;

    const half_t* Qp = Qh + ((size_t)bh * LQ + qrow0) * HDD;
    const half_t* Kp = Kh + (size_t)bh * LKV * HDD;
    const half_t* Vp = Vh + (size_t)bh * LKV * HDD;

    int mq = lane & 15, kb = (lane >> 4) * 8, g2 = (lane >> 4) * 8;
    uint32_t ldsK = (uint32_t)(uintptr_t)&Kbuf[0][0];

    // Q fragments (pre-scaled by D^-0.5), d-chunks 0..31 and 32..63
    v16h aq[2];
    #pragma unroll
    for (int c = 0; c < 2; ++c) {
        const half_t* qr = Qp + (size_t)mq * HDD + c * 32;
        v8h lo = *reinterpret_cast<const v8h*>(qr + 2 * kb);
        v8h hi = *reinterpret_cast<const v8h*>(qr + 2 * kb + 8);
        #pragma unroll
        for (int i = 0; i < 8; ++i) {
            aq[c][i]     = (half_t)((float)lo[i] * QK_SCALE);
            aq[c][i + 8] = (half_t)((float)hi[i] * QK_SCALE);
        }
    }

    float mrow[8], lrow[8];
    #pragma unroll
    for (int r = 0; r < 8; ++r) { mrow[r] = -__builtin_inff(); lrow[r] = 0.0f; }
    v8f o[4] = {};

    for (int kv0 = 0; kv0 < Tkv; kv0 += 32) {
        // ---- cooperative staging (no divergence: 128 threads x 2 chunks each)
        // K: linear async copy global->LDS (4KB = 256 x 16B)
        for (int c = tid; c < 256; c += 128) {
            uint64_t ga = (uint64_t)(uintptr_t)(Kp + (size_t)kv0 * HDD) + (uint64_t)c * 16u;
            uint32_t la = ldsK + (uint32_t)c * 16u;
            asm volatile("global_load_async_to_lds_b128 %0, %1, off"
                         :: "v"(la), "v"(ga) : "memory");
        }
        // V: coalesced 16B global reads, transposed scalar DS stores
        for (int c = tid; c < 256; c += 128) {
            int kv = c >> 3, d0 = (c & 7) * 8;
            v8h vrow = *reinterpret_cast<const v8h*>(Vp + (size_t)(kv0 + kv) * HDD + d0);
            #pragma unroll
            for (int i = 0; i < 8; ++i) Vt[d0 + i][kv] = vrow[i];
        }
        asm volatile("s_wait_asynccnt 0x0" ::: "memory");
        __syncthreads();

        // ---- S tiles from LDS K (contiguous DS reads)
        v8f s0 = {}, s1 = {};
        #pragma unroll
        for (int c = 0; c < 2; ++c) {
            v16h b0 = *reinterpret_cast<const v16h*>(&Kbuf[lane & 15][c * 32 + (lane >> 4) * 16]);
            v16h b1 = *reinterpret_cast<const v16h*>(&Kbuf[16 + (lane & 15)][c * 32 + (lane >> 4) * 16]);
            s0 = __builtin_amdgcn_wmma_f32_16x16x32_f16(false, aq[c], false, b0, (short)0, s0, false, false);
            s1 = __builtin_amdgcn_wmma_f32_16x16x32_f16(false, aq[c], false, b1, (short)0, s1, false, false);
        }
        // ---- online softmax update (row = r + g2; N striped across 16-lane halves)
        #pragma unroll
        for (int r = 0; r < 8; ++r) {
            float mloc = fmaxf(s0[r], s1[r]);
            #pragma unroll
            for (int off = 1; off < 16; off <<= 1)
                mloc = fmaxf(mloc, __shfl_xor(mloc, off, 32));
            float mnew = fmaxf(mrow[r], mloc);
            float alpha = __expf(mrow[r] - mnew);
            float p0 = __expf(s0[r] - mnew);
            float p1 = __expf(s1[r] - mnew);
            float sloc = p0 + p1;
            #pragma unroll
            for (int off = 1; off < 16; off <<= 1)
                sloc += __shfl_xor(sloc, off, 32);
            lrow[r] = lrow[r] * alpha + sloc;
            mrow[r] = mnew;
            #pragma unroll
            for (int dt = 0; dt < 4; ++dt) o[dt][r] *= alpha;
            s0[r] = p0; s1[r] = p1;
        }
        // ---- P through per-wave LDS slab (DS ops in-order within a wave)
        #pragma unroll
        for (int r = 0; r < 8; ++r) {
            int row = r + g2;
            Plds[wid][row][lane & 15]        = (half_t)s0[r];
            Plds[wid][row][16 + (lane & 15)] = (half_t)s1[r];
        }
        v8h plo = *reinterpret_cast<const v8h*>(&Plds[wid][mq][2 * kb]);
        v8h phi = *reinterpret_cast<const v8h*>(&Plds[wid][mq][2 * kb + 8]);
        v16h pf;
        #pragma unroll
        for (int i = 0; i < 8; ++i) { pf[i] = plo[i]; pf[i + 8] = phi[i]; }
        // ---- O += P * V  (V^T B-fragment = one contiguous 32B DS read)
        #pragma unroll
        for (int dt = 0; dt < 4; ++dt) {
            v16h vb = *reinterpret_cast<const v16h*>(&Vt[dt * 16 + (lane & 15)][(lane >> 4) * 16]);
            o[dt] = __builtin_amdgcn_wmma_f32_16x16x32_f16(false, pf, false, vb, (short)0, o[dt], false, false);
        }
        __syncthreads();   // protect K/V buffers before next staging round
    }
    // ---- normalize, write merged head layout
    #pragma unroll
    for (int r = 0; r < 8; ++r) {
        float inv = 1.0f / lrow[r];
        int row = qrow0 + r + g2;
        #pragma unroll
        for (int dt = 0; dt < 4; ++dt)
            xattn[((size_t)bq * LQ + row) * DIM + hh * HDD + dt * 16 + (lane & 15)] =
                (half_t)(o[dt][r] * inv);
    }
}

// ---------------------------------------------------------------------------
// Final projection: out(M x 768, f32) = xattn * w_proj^T + b_proj (pipelined)
// ---------------------------------------------------------------------------
__global__ void gemm_out(const half_t* __restrict__ A, const half_t* __restrict__ Wt,
                         const float* __restrict__ bias, float* __restrict__ out, int M) {
    const int NG = DIM / 64;
    int wave = blockIdx.x * (blockDim.x >> 5) + (threadIdx.x >> 5);
    int lane = threadIdx.x & 31;
    int mt = wave / NG, ng = wave % NG;
    if (mt >= M / 16) return;
    int row0 = mt * 16, n0 = ng * 64;
    int mloc = lane & 15, kb = (lane >> 4) * 8, koff = (lane >> 4) * 16;
    const half_t* arow = A + (size_t)(row0 + mloc) * DIM;
    const half_t* wrow[4];
    #pragma unroll
    for (int nt = 0; nt < 4; ++nt)
        wrow[nt] = Wt + (size_t)(n0 + nt * 16 + (lane & 15)) * DIM + koff;

    v8f acc[4] = {};
    v16h a_c = make_a_frag(arow, kb);
    v16h b_c[4];
    #pragma unroll
    for (int nt = 0; nt < 4; ++nt)
        b_c[nt] = *reinterpret_cast<const v16h*>(wrow[nt]);

    for (int k0 = 0; k0 < DIM; k0 += 32) {
        int kn = (k0 + 32 < DIM) ? (k0 + 32) : k0;
        v16h a_n = make_a_frag(arow + kn, kb);
        v16h b_n[4];
        #pragma unroll
        for (int nt = 0; nt < 4; ++nt)
            b_n[nt] = *reinterpret_cast<const v16h*>(wrow[nt] + kn);
        #pragma unroll
        for (int nt = 0; nt < 4; ++nt)
            acc[nt] = __builtin_amdgcn_wmma_f32_16x16x32_f16(
                false, a_c, false, b_c[nt], (short)0, acc[nt], false, false);
        a_c = a_n;
        #pragma unroll
        for (int nt = 0; nt < 4; ++nt) b_c[nt] = b_n[nt];
    }
    int g2 = (lane >> 4) * 8;
    #pragma unroll
    for (int nt = 0; nt < 4; ++nt) {
        #pragma unroll
        for (int r = 0; r < 8; ++r) {
            int row = row0 + r + g2;
            int col = n0 + nt * 16 + (lane & 15);
            out[(size_t)row * DIM + col] = acc[nt][r] + bias[col];
        }
    }
}

// ---------------------------------------------------------------------------
extern "C" void kernel_launch(void* const* d_in, const int* in_sizes, int n_in,
                              void* d_out, int out_size, void* d_ws, size_t ws_size,
                              hipStream_t stream) {
    (void)in_sizes; (void)n_in; (void)out_size; (void)ws_size;
    const float* x      = (const float*)d_in[0];
    const float* cqw    = (const float*)d_in[1];
    const float* qg     = (const float*)d_in[2];
    const float* qb     = (const float*)d_in[3];
    const float* qm     = (const float*)d_in[4];
    const float* qv     = (const float*)d_in[5];
    const float* ckw    = (const float*)d_in[6];
    const float* kg     = (const float*)d_in[7];
    const float* kb_    = (const float*)d_in[8];
    const float* km     = (const float*)d_in[9];
    const float* kv_    = (const float*)d_in[10];
    const float* cvw    = (const float*)d_in[11];
    const float* vg     = (const float*)d_in[12];
    const float* vb     = (const float*)d_in[13];
    const float* vm     = (const float*)d_in[14];
    const float* vv     = (const float*)d_in[15];
    const float* wq     = (const float*)d_in[16];
    const float* wk     = (const float*)d_in[17];
    const float* wv     = (const float*)d_in[18];
    const float* wp     = (const float*)d_in[19];
    const float* bp     = (const float*)d_in[20];
    float* out = (float*)d_out;

    char* base = (char*)d_ws;
    size_t off = 0;
    auto carve = [&](size_t bytes) -> void* {
        void* p = base + off;
        off = (off + bytes + 255) & ~(size_t)255;
        return p;
    };
    const size_t SQ = (size_t)NB * LQ  * DIM * sizeof(half_t);
    const size_t SK = (size_t)NB * LKV * DIM * sizeof(half_t);
    const size_t SW = (size_t)DIM * DIM * sizeof(half_t);
    half_t* qc  = (half_t*)carve(SQ);
    half_t* kc  = (half_t*)carve(SK);
    half_t* vc  = (half_t*)carve(SK);
    half_t* Qh  = (half_t*)carve(SQ);
    half_t* Kh  = (half_t*)carve(SK);
    half_t* Vh  = (half_t*)carve(SK);
    half_t* wqh = (half_t*)carve(SW);
    half_t* wkh = (half_t*)carve(SW);
    half_t* wvh = (half_t*)carve(SW);
    half_t* wph = (half_t*)carve(SW);
    half_t* xattn = qc;   // qc dead after Q projection; alias it

    {
        int n = DIM * DIM;
        dim3 g((n + 255) / 256), b(256);
        cvt_f32_f16<<<g, b, 0, stream>>>(wq, wqh, n);
        cvt_f32_f16<<<g, b, 0, stream>>>(wk, wkh, n);
        cvt_f32_f16<<<g, b, 0, stream>>>(wv, wvh, n);
        cvt_f32_f16<<<g, b, 0, stream>>>(wp, wph, n);
    }
    convbn_s1<<<dim3(NB * LQ), dim3(256), 0, stream>>>(x, cqw, qg, qb, qm, qv, qc);
    convbn_s2<<<dim3(NB * LKV), dim3(256), 0, stream>>>(x, ckw, kg, kb_, km, kv_,
                                                        cvw, vg, vb, vm, vv, kc, vc);
    {
        int Mq = NB * LQ;
        int Mk = NB * LKV;
        gemm_qkv<<<dim3((Mq / 16) * 12 / 8), dim3(256), 0, stream>>>(qc, wqh, Qh, Mq, LQ);
        gemm_qkv<<<dim3((Mk / 16) * 12 / 8), dim3(256), 0, stream>>>(kc, wkh, Kh, Mk, LKV);
        gemm_qkv<<<dim3((Mk / 16) * 12 / 8), dim3(256), 0, stream>>>(vc, wvh, Vh, Mk, LKV);
    }
    attn_flash<<<dim3(NB * NHD, 512 / 64), dim3(128), 0, stream>>>(Qh, Kh, Vh, xattn, 0,   MTKV);
    attn_flash<<<dim3(NB * NHD, 1024 / 64), dim3(128), 0, stream>>>(Qh, Kh, Vh, xattn, 512, LKV);
    gemm_out<<<dim3(((NB * LQ) / 16) * 12 / 8), dim3(256), 0, stream>>>(xattn, wph, bp, out, NB * LQ);
}
